// MultiHeadAttention_60756607369721
// MI455X (gfx1250) — matrix operand order repaired and checked by reading, not compiled, not run on hardware
//
#include <hip/hip_runtime.h>
#include <hip/hip_bf16.h>
#include <stdint.h>

// ---------------------------------------------------------------------------
// Types for CDNA5 WMMA (wave32): v_wmma_f32_16x16x32_bf16
// ---------------------------------------------------------------------------
typedef __attribute__((ext_vector_type(16))) __bf16       v16bf;
typedef __attribute__((ext_vector_type(8)))  float        v8f;
typedef __attribute__((ext_vector_type(8)))  unsigned int v8u;

#define B_  2
#define T_  2048
#define D_  1024
#define H_  16
#define HD_ 64
#define M_  (B_ * T_)   // 4096 rows

// ---------------------------------------------------------------------------
// Helpers
// ---------------------------------------------------------------------------
__device__ __forceinline__ unsigned short f32_to_bf16(float f) {
  unsigned int u = __builtin_bit_cast(unsigned int, f);
  unsigned int r = 0x7FFFu + ((u >> 16) & 1u);   // round-to-nearest-even
  return (unsigned short)((u + r) >> 16);
}

// A-fragment (16x32, 16-bit): lane half h, VGPR v -> K pair base
__device__ __forceinline__ int kA_off(int v, int half) {
  return (half << 4) + (v << 1);
}
// B-fragment (32x16, 16-bit): lane half h, VGPR v -> K pair base
__device__ __forceinline__ int kB_off(int v, int half) {
  return (half << 4) + (v << 1);
}

// Load A fragment: p = pointer to this lane's row (bf16, contiguous K)
__device__ __forceinline__ v16bf load_fragA(const unsigned short* p, int half) {
  v8u t;
#pragma unroll
  for (int v = 0; v < 8; ++v)
    t[v] = *(const unsigned int*)(p + kA_off(v, half));
  return __builtin_bit_cast(v16bf, t);
}
// Load B fragment where source row n holds K contiguously (i.e. B = Wt / Kt)
__device__ __forceinline__ v16bf load_fragB_rowmajorK(const unsigned short* p, int half) {
  v8u t;
#pragma unroll
  for (int v = 0; v < 8; ++v)
    t[v] = *(const unsigned int*)(p + kB_off(v, half));
  return __builtin_bit_cast(v16bf, t);
}

__device__ __forceinline__ v8f wmma_bf16(v16bf a, v16bf b, v8f c) {
  return __builtin_amdgcn_wmma_f32_16x16x32_bf16(false, a, false, b, (short)0, c,
                                                 false, false);
}

// gfx1250 async DMA: global (16B) -> LDS, tracked by ASYNCcnt.
// LDS aperture flat pointers carry the wave-relative LDS offset in addr[31:0]
// (ISA 10.2 aperture mapping), so truncation yields the VDST operand.
__device__ __forceinline__ void async_ld16(const unsigned short* g,
                                           unsigned short* lds) {
  unsigned loff = (unsigned)(unsigned long long)lds;
  asm volatile("global_load_async_to_lds_b128 %0, %1, off"
               :: "v"(loff), "v"((unsigned long long)g)
               : "memory");
}
__device__ __forceinline__ void wait_async0() {
  asm volatile("s_wait_asynccnt 0x0" ::: "memory");
}

// row-wise reductions across the 16 lanes of a wave half
__device__ __forceinline__ float rowmax16(float x) {
#pragma unroll
  for (int m = 1; m < 16; m <<= 1) x = fmaxf(x, __shfl_xor(x, m, 32));
  return x;
}
__device__ __forceinline__ float rowsum16(float x) {
#pragma unroll
  for (int m = 1; m < 16; m <<= 1) x += __shfl_xor(x, m, 32);
  return x;
}

// ---------------------------------------------------------------------------
// fp32 -> bf16 conversion
// ---------------------------------------------------------------------------
__global__ void cvt_f32_to_bf16_kernel(const float* __restrict__ in,
                                       unsigned short* __restrict__ out, int n) {
  int i = blockIdx.x * blockDim.x + threadIdx.x;
  if (i < n) out[i] = f32_to_bf16(in[i]);
}

// ---------------------------------------------------------------------------
// GEMM: C[M,N] = (A[M,K] @ W[N,K]^T + bias) * scale
// 256 threads = 8 waves; block tile 128x128; wave tile 32x64 (2x4 WMMA tiles)
// ---------------------------------------------------------------------------
__global__ __launch_bounds__(256) void gemm_bf16_kernel(
    const unsigned short* __restrict__ A, const unsigned short* __restrict__ W,
    const float* __restrict__ bias, float scale,
    unsigned short* __restrict__ outB, float* __restrict__ outF,
    int M, int N, int K) {
  const int l    = threadIdx.x & 31;
  const int wv   = threadIdx.x >> 5;     // 0..7
  const int wm   = wv >> 1;              // 0..3
  const int wn   = wv & 1;               // 0..1
  const int m16  = l & 15;
  const int half = l >> 4;

  const int rowA0 = blockIdx.y * 128 + wm * 32;   // wave: 32 rows
  const int colB0 = blockIdx.x * 128 + wn * 64;   // wave: 64 cols

  v8f acc[2][4] = {};

  for (int kk = 0; kk < K; kk += 32) {
    if (kk + 32 < K) {
      __builtin_prefetch(A + (size_t)(rowA0 + m16) * K + kk + 32, 0, 1);
      __builtin_prefetch(W + (size_t)(colB0 + m16) * K + kk + 32, 0, 1);
    }
    v16bf a[2];
#pragma unroll
    for (int i = 0; i < 2; ++i)
      a[i] = load_fragA(A + (size_t)(rowA0 + i * 16 + m16) * K + kk, half);
    v16bf b[4];
#pragma unroll
    for (int j = 0; j < 4; ++j)
      b[j] = load_fragB_rowmajorK(W + (size_t)(colB0 + j * 16 + m16) * K + kk, half);
#pragma unroll
    for (int i = 0; i < 2; ++i)
#pragma unroll
      for (int j = 0; j < 4; ++j)
        acc[i][j] = wmma_bf16(a[i], b[j], acc[i][j]);
  }

  // epilogue: C layout -> row = r + 8*half, col = l&15
#pragma unroll
  for (int i = 0; i < 2; ++i) {
#pragma unroll
    for (int j = 0; j < 4; ++j) {
      const int col = colB0 + j * 16 + m16;
      const float bval = bias ? bias[col] : 0.0f;
#pragma unroll
      for (int r = 0; r < 8; ++r) {
        const int row = rowA0 + i * 16 + r + 8 * half;
        const float v = (acc[i][j][r] + bval) * scale;
        if (outF) outF[(size_t)row * N + col] = v;
        else      outB[(size_t)row * N + col] = f32_to_bf16(v);
      }
    }
  }
}

// ---------------------------------------------------------------------------
// Flash attention (causal), bf16 in/out, f32 online softmax.
// 128 threads = 4 waves; wave owns 16 query rows; block owns 64 rows.
// K/V tiles (32 keys x 64 d) are DMA'd into LDS with async loads, double-
// buffered so the next tile's DMA overlaps this tile's WMMA + softmax.
// Q/K pre-scaled by HD^-0.25 each (done in projection GEMMs).
// ---------------------------------------------------------------------------
#define KROW 72   // padded row: 144 B (16B aligned; stride 36 dwords -> no bank conflicts)
#define PROW 40

__global__ __launch_bounds__(128) void flash_attn_kernel(
    const unsigned short* __restrict__ Q, const unsigned short* __restrict__ Kb,
    const unsigned short* __restrict__ V, unsigned short* __restrict__ O) {
  __shared__ __align__(16) unsigned short Kt[2][32][KROW];
  __shared__ __align__(16) unsigned short Vt[2][32][KROW];
  __shared__ __align__(16) unsigned short Plds[4][16 * PROW];

  const int tid  = threadIdx.x;
  const int l    = tid & 31;
  const int w    = tid >> 5;   // 0..3
  const int m16  = l & 15;
  const int half = l >> 4;

  const int bh = blockIdx.y;           // b*H + h
  const int b  = bh >> 4;
  const int h  = bh & 15;
  const int qTile = blockIdx.x * 64;
  const int q0    = qTile + w * 16;

  const size_t headOff = ((size_t)b * T_) * D_ + (size_t)h * HD_;

  // cooperative staging: 32 rows x 8 x 16B chunks -> 2 chunks/thread/tensor
  const int sr = tid >> 3;             // 0..15 (and sr+16)
  const int sc = (tid & 7) * 8;        // element column of 16B chunk

  // Q fragments: 16 rows x 64 d  -> two 16x32 A fragments
  v16bf qa[2];
#pragma unroll
  for (int c = 0; c < 2; ++c)
    qa[c] = load_fragA(Q + headOff + (size_t)(q0 + m16) * D_ + c * 32, half);

  v8f o[4] = {};
  float mrow[8], lrow[8];
#pragma unroll
  for (int r = 0; r < 8; ++r) { mrow[r] = -3.0e38f; lrow[r] = 0.0f; }

  const int nkt = (qTile + 64) / 32;   // causal bound for the whole block

  // prologue: stage tile 0 into buffer 0
  {
    const unsigned short* gK = Kb + headOff;
    const unsigned short* gV = V  + headOff;
    async_ld16(gK + (size_t)sr * D_ + sc,        &Kt[0][sr][sc]);
    async_ld16(gK + (size_t)(sr + 16) * D_ + sc, &Kt[0][sr + 16][sc]);
    async_ld16(gV + (size_t)sr * D_ + sc,        &Vt[0][sr][sc]);
    async_ld16(gV + (size_t)(sr + 16) * D_ + sc, &Vt[0][sr + 16][sc]);
  }

  for (int kt = 0; kt < nkt; ++kt) {
    const int buf = kt & 1;
    wait_async0();        // own transfers done
    __syncthreads();      // everyone's transfers visible; prev readers drained

    if (kt + 1 < nkt) {   // overlap next tile's DMA with this tile's math
      const int nb = (kt + 1) & 1;
      const size_t base = headOff + (size_t)((kt + 1) * 32) * D_;
      async_ld16(Kb + base + (size_t)sr * D_ + sc,        &Kt[nb][sr][sc]);
      async_ld16(Kb + base + (size_t)(sr + 16) * D_ + sc, &Kt[nb][sr + 16][sc]);
      async_ld16(V  + base + (size_t)sr * D_ + sc,        &Vt[nb][sr][sc]);
      async_ld16(V  + base + (size_t)(sr + 16) * D_ + sc, &Vt[nb][sr + 16][sc]);
    }

    const int key0 = kt * 32;

    // ---- S = Q @ K^T : two 16x16 tiles over 32 keys (K from LDS) -------
    v8f s[2];
#pragma unroll
    for (int t = 0; t < 2; ++t) {
      const unsigned short* kr = &Kt[buf][t * 16 + m16][0];
      v8u t0, t1;
#pragma unroll
      for (int v = 0; v < 8; ++v) {
        t0[v] = *(const unsigned int*)(kr + kB_off(v, half));
        t1[v] = *(const unsigned int*)(kr + 32 + kB_off(v, half));
      }
      v8f z = {};
      z = wmma_bf16(qa[0], __builtin_bit_cast(v16bf, t0), z);
      z = wmma_bf16(qa[1], __builtin_bit_cast(v16bf, t1), z);
      s[t] = z;
    }

    // ---- causal mask (reference adds -1e9 on strict upper triangle) ---
#pragma unroll
    for (int t = 0; t < 2; ++t) {
      const int key = key0 + t * 16 + m16;
#pragma unroll
      for (int r = 0; r < 8; ++r) {
        const int qrow = q0 + r + 8 * half;
        if (key > qrow) s[t][r] += -1.0e9f;
      }
    }

    // ---- online softmax update ----------------------------------------
#pragma unroll
    for (int r = 0; r < 8; ++r) {
      float tmax = rowmax16(fmaxf(s[0][r], s[1][r]));
      float nm   = fmaxf(mrow[r], tmax);
      float alpha = __expf(mrow[r] - nm);
      float p0 = __expf(s[0][r] - nm);
      float p1 = __expf(s[1][r] - nm);
      s[0][r] = p0;
      s[1][r] = p1;
      lrow[r] = lrow[r] * alpha + rowsum16(p0 + p1);
      mrow[r] = nm;
#pragma unroll
      for (int j = 0; j < 4; ++j) o[j][r] *= alpha;
    }

    // ---- re-swizzle P (C layout) -> A fragment via per-wave LDS -------
#pragma unroll
    for (int t = 0; t < 2; ++t)
#pragma unroll
      for (int r = 0; r < 8; ++r)
        Plds[w][(r + 8 * half) * PROW + t * 16 + m16] = f32_to_bf16(s[t][r]);

    v8u pt;
#pragma unroll
    for (int v = 0; v < 8; ++v)
      pt[v] = *(const unsigned int*)(&Plds[w][m16 * PROW + kA_off(v, half)]);
    v16bf pa = __builtin_bit_cast(v16bf, pt);

    // ---- O += P(16x32) @ V(32x64) : 4 WMMAs (V from LDS) --------------
#pragma unroll
    for (int j = 0; j < 4; ++j) {
      v8u tv;
#pragma unroll
      for (int v = 0; v < 8; ++v) {
        const int k0 = kB_off(v, half);
        unsigned int lo = Vt[buf][k0][j * 16 + m16];
        unsigned int hi = Vt[buf][k0 + 1][j * 16 + m16];
        tv[v] = lo | (hi << 16);
      }
      o[j] = wmma_bf16(pa, __builtin_bit_cast(v16bf, tv), o[j]);
    }
  }

  // ---- finalize: O /= l, store bf16 [B*T, D] --------------------------
#pragma unroll
  for (int r = 0; r < 8; ++r) {
    const float inv = 1.0f / lrow[r];
    const size_t base = headOff + (size_t)(q0 + r + 8 * half) * D_;
#pragma unroll
    for (int j = 0; j < 4; ++j)
      O[base + j * 16 + m16] = f32_to_bf16(o[j][r] * inv);
  }
}

// ---------------------------------------------------------------------------
// Host-side launch
// ---------------------------------------------------------------------------
extern "C" void kernel_launch(void* const* d_in, const int* in_sizes, int n_in,
                              void* d_out, int out_size, void* d_ws, size_t ws_size,
                              hipStream_t stream) {
  (void)in_sizes; (void)n_in; (void)out_size; (void)ws_size;

  const float* x  = (const float*)d_in[0];
  // d_in[1] = mask (unused: causal mask computed analytically)
  const float* Wq = (const float*)d_in[2];
  const float* bq = (const float*)d_in[3];
  const float* Wk = (const float*)d_in[4];
  const float* Wv = (const float*)d_in[5];
  const float* bv = (const float*)d_in[6];
  const float* Wo = (const float*)d_in[7];
  const float* bo = (const float*)d_in[8];
  float* out = (float*)d_out;

  // workspace layout (bf16 = unsigned short), total 48 MB
  unsigned short* ws  = (unsigned short*)d_ws;
  unsigned short* xb  = ws;                          // [M, D]
  unsigned short* wqb = xb  + (size_t)M_ * D_;       // [D, D]
  unsigned short* wkb = wqb + (size_t)D_ * D_;
  unsigned short* wvb = wkb + (size_t)D_ * D_;
  unsigned short* wob = wvb + (size_t)D_ * D_;
  unsigned short* qb  = wob + (size_t)D_ * D_;       // [M, D]
  unsigned short* kb  = qb  + (size_t)M_ * D_;
  unsigned short* vb  = kb  + (size_t)M_ * D_;
  unsigned short* ab  = vb  + (size_t)M_ * D_;       // attention output [M, D]

  const int nMD = M_ * D_;
  const int nDD = D_ * D_;
  cvt_f32_to_bf16_kernel<<<(nMD + 255) / 256, 256, 0, stream>>>(x,  xb,  nMD);
  cvt_f32_to_bf16_kernel<<<(nDD + 255) / 256, 256, 0, stream>>>(Wq, wqb, nDD);
  cvt_f32_to_bf16_kernel<<<(nDD + 255) / 256, 256, 0, stream>>>(Wk, wkb, nDD);
  cvt_f32_to_bf16_kernel<<<(nDD + 255) / 256, 256, 0, stream>>>(Wv, wvb, nDD);
  cvt_f32_to_bf16_kernel<<<(nDD + 255) / 256, 256, 0, stream>>>(Wo, wob, nDD);

  const float s = 0.35355339059327379f;  // HD^-0.25 = 64^-0.25
  dim3 gg(D_ / 128, M_ / 128);           // (8, 32)

  // q = (x@Wq^T + bq)*s ; k = (x@Wk^T)*s ; v = x@Wv^T + bv
  gemm_bf16_kernel<<<gg, 256, 0, stream>>>(xb, wqb, bq, s,     qb, nullptr, M_, D_, D_);
  gemm_bf16_kernel<<<gg, 256, 0, stream>>>(xb, wkb, nullptr, s, kb, nullptr, M_, D_, D_);
  gemm_bf16_kernel<<<gg, 256, 0, stream>>>(xb, wvb, bv, 1.0f,  vb, nullptr, M_, D_, D_);

  // causal flash attention (async-LDS double-buffered K/V tiles)
  flash_attn_kernel<<<dim3(T_ / 64, B_ * H_), 128, 0, stream>>>(qb, kb, vb, ab);

  // out = att @ Wo^T + bo   (fp32 output)
  gemm_bf16_kernel<<<gg, 256, 0, stream>>>(ab, wob, bo, 1.0f, nullptr, out, M_, D_, D_);
}
